// SelfAttention_11458972746290
// MI455X (gfx1250) — matrix operand order repaired and checked by reading, not compiled, not run on hardware
//
#include <hip/hip_runtime.h>
#include <hip/hip_bf16.h>

// gfx1250 wave32 WMMA types
typedef __attribute__((ext_vector_type(16))) _Float16 v16h;
typedef __attribute__((ext_vector_type(4)))  _Float16 v4h;
typedef __attribute__((ext_vector_type(8)))  float    v8f;
typedef __attribute__((ext_vector_type(4)))  int      v4i;

#define TT   2048      // sequence length
#define CC   1024      // embed dim
#define HH   16        // heads
#define DD   64        // head dim
#define BB   4         // batch
#define ROWS (BB*TT)   // 8192

#define AS1 __attribute__((address_space(1)))
#define AS3 __attribute__((address_space(3)))

// ---------------------------------------------------------------------------
// CDNA5 async global->LDS staging (ASYNCcnt path), with portable fallback.
// Async loads retire in order per wave, so waiting ASYNCcnt<=1 completes the
// older (current) panel while the prefetch for the next panel stays in flight.
// ---------------------------------------------------------------------------
#if defined(__gfx1250__) && __has_builtin(__builtin_amdgcn_global_load_async_to_lds_b128)
#define HAVE_ASYNC_LDS 1
#define ASYNC_LDS_B128(gsrc, ldst)                                            \
  __builtin_amdgcn_global_load_async_to_lds_b128(                             \
      (AS1 v4i*)(void*)(gsrc), (AS3 v4i*)(void*)(ldst), 0, 0)
#define ASYNC_WAIT_LE0() asm volatile("s_wait_asynccnt 0x0" ::: "memory")
#define ASYNC_WAIT_LE1() asm volatile("s_wait_asynccnt 0x1" ::: "memory")
#else
#define HAVE_ASYNC_LDS 0
#define ASYNC_LDS_B128(gsrc, ldst)                                            \
  (*(uint4*)(ldst) = *(const uint4*)(gsrc))
#define ASYNC_WAIT_LE0() ((void)0)
#define ASYNC_WAIT_LE1() ((void)0)
#endif

__device__ __forceinline__ v8f wmma16(v16h a, v16h b, v8f c) {
  // D = A(16x32 f16) x B(32x16 f16) + C(16x16 f32)
  return __builtin_amdgcn_wmma_f32_16x16x32_f16(false, a, false, b, (short)0, c,
                                                false, false);
}

__device__ __forceinline__ float half16_max(float v) {
  v = fmaxf(v, __shfl_xor(v, 1, 32));
  v = fmaxf(v, __shfl_xor(v, 2, 32));
  v = fmaxf(v, __shfl_xor(v, 4, 32));
  v = fmaxf(v, __shfl_xor(v, 8, 32));
  return v;
}
__device__ __forceinline__ float half16_sum(float v) {
  v += __shfl_xor(v, 1, 32);
  v += __shfl_xor(v, 2, 32);
  v += __shfl_xor(v, 4, 32);
  v += __shfl_xor(v, 8, 32);
  return v;
}

// ---------------------------------------------------------------------------
// Convert x (f32) -> xh (f16), elementwise, vectorized.
// grid = ROWS*CC/4/256, block = 256
// ---------------------------------------------------------------------------
__global__ __launch_bounds__(256)
void cvt_x_kernel(const float* __restrict__ x, _Float16* __restrict__ xh) {
  const size_t i = ((size_t)blockIdx.x * 256 + threadIdx.x) * 4;
  const float4 f = *(const float4*)(x + i);
  v4h h;
  h[0] = (_Float16)f.x; h[1] = (_Float16)f.y;
  h[2] = (_Float16)f.z; h[3] = (_Float16)f.w;
  *(v4h*)(xh + i) = h;
}

// ---------------------------------------------------------------------------
// Convert + transpose weights: W f32 [k][n] -> Wt f16 [n][k].
// LDS-tiled so both global accesses are coalesced.
// grid = (CC/32, CC/32, 4), block = 256
// ---------------------------------------------------------------------------
__global__ __launch_bounds__(256)
void cvt_wt_kernel(const float* __restrict__ W0, const float* __restrict__ W1,
                   const float* __restrict__ W2, const float* __restrict__ W3,
                   _Float16* __restrict__ T0, _Float16* __restrict__ T1,
                   _Float16* __restrict__ T2, _Float16* __restrict__ T3) {
  const float* W; _Float16* Wt;
  switch (blockIdx.z) {
    case 0:  W = W0; Wt = T0; break;
    case 1:  W = W1; Wt = T1; break;
    case 2:  W = W2; Wt = T2; break;
    default: W = W3; Wt = T3; break;
  }
  __shared__ _Float16 tl[32][33];
  const int nt = blockIdx.x * 32;        // n tile base
  const int kt = blockIdx.y * 32;        // k tile base
  const int lx = threadIdx.x & 31;
  const int ly = threadIdx.x >> 5;       // 0..7
#pragma unroll
  for (int j = 0; j < 32; j += 8)
    tl[ly + j][lx] = (_Float16)W[(size_t)(kt + ly + j) * CC + nt + lx];
  __syncthreads();
#pragma unroll
  for (int j = 0; j < 32; j += 8)
    Wt[(size_t)(nt + ly + j) * CC + kt + lx] = tl[lx][ly + j];
}

// ---------------------------------------------------------------------------
// GEMM core: block = 256 (8 waves). Block tile = 128 rows x 64 cols.
// Each wave computes 16 rows x 64 cols (4 accumulators, 4 WMMA per K-step).
// B panel (64 cols x 32 k of Wt) double-buffered in LDS: the async DMA for
// K-chunk i+1 is issued before computing chunk i (s_wait_asynccnt <= 1),
// overlapping global->LDS traffic with the WMMA pipeline.
// ---------------------------------------------------------------------------
#define PANEL (64 * 32)

__device__ __forceinline__ void gemm_core(const _Float16* __restrict__ A,
                                          const _Float16* __restrict__ Wt,
                                          _Float16* btile /* [2][PANEL] */,
                                          int row0, int col0, v8f acc[4]) {
  const int tid  = threadIdx.x;
  const int lane = tid & 31;
  const int m  = lane & 15;
  const int kh = lane >> 4;
  const int n  = lane & 15;
  const int sn = tid >> 2;        // 0..63 : panel column staged by this thread
  const int sk = (tid & 3) * 8;   // 0,8,16,24 : halves (16B) within k-chunk
  const _Float16* gbase = Wt + (size_t)(col0 + sn) * CC + sk;
  const int soff = sn * 32 + sk;

  // prologue: stage panel 0 into buffer 0
  ASYNC_LDS_B128(gbase, &btile[soff]);

  for (int k0 = 0; k0 < CC; k0 += 32) {
    _Float16* cur = btile + ((k0 >> 5) & 1) * PANEL;
    if (k0 + 32 < CC) {
      // prefetch next panel into the other buffer (its readers finished
      // before the end-of-iteration barrier of the previous iteration)
      _Float16* nxt = btile + (((k0 >> 5) + 1) & 1) * PANEL;
      ASYNC_LDS_B128(gbase + k0 + 32, &nxt[soff]);
      ASYNC_WAIT_LE1();   // current panel complete; prefetch stays in flight
    } else {
      ASYNC_WAIT_LE0();
    }
    __syncthreads();      // current panel visible to all waves

    // A operand (K = 16*kh+i, matching B)
    v16h a;
    const _Float16* ap = A + (size_t)(row0 + m) * CC + k0;
#pragma unroll
    for (int i = 0; i < 8; ++i) a[i]     = ap[kh * 16 + i];
#pragma unroll
    for (int i = 0; i < 8; ++i) a[8 + i] = ap[kh * 16 + 8 + i];

#pragma unroll
    for (int nt = 0; nt < 4; ++nt) {
      v16h bm;
      const _Float16* lp = &cur[(nt * 16 + n) * 32 + kh * 16];
#pragma unroll
      for (int i = 0; i < 16; ++i) bm[i] = lp[i];
      acc[nt] = wmma16(a, bm, acc[nt]);
    }
    __syncthreads();      // all reads of 'cur' done before it is re-staged
  }
}

// ---------------------------------------------------------------------------
// Kernel: QKV projections.  Q,K stored f16 [B,H,T,D]; V stored f16 [B,H,D,T]
// (transposed so the PV B-operand in attention is contiguous).
// grid = (CC/64, ROWS/128, 3), block = 256
// ---------------------------------------------------------------------------
__global__ __launch_bounds__(256)
void qkv_proj_kernel(const _Float16* __restrict__ xh,
                     const _Float16* __restrict__ Wtq, const float* __restrict__ bq,
                     const _Float16* __restrict__ Wtk, const float* __restrict__ bk,
                     const _Float16* __restrict__ Wtv, const float* __restrict__ bv,
                     _Float16* __restrict__ Qo, _Float16* __restrict__ Ko,
                     _Float16* __restrict__ Vto) {
  const _Float16* Wt; const float* bias;
  if (blockIdx.z == 0)      { Wt = Wtq; bias = bq; }
  else if (blockIdx.z == 1) { Wt = Wtk; bias = bk; }
  else                      { Wt = Wtv; bias = bv; }

  __shared__ _Float16 btile[2 * PANEL];
  const int wave = threadIdx.x >> 5;
  const int lane = threadIdx.x & 31;
  const int row0 = blockIdx.y * 128 + wave * 16;
  const int col0 = blockIdx.x * 64;
  const int kh = lane >> 4;
  const int n  = lane & 15;
  const int mb = 8 * kh;              // C/D layout: row = r + 8*(lane>=16)

  v8f acc[4] = {{}, {}, {}, {}};
  gemm_core(xh, Wt, btile, row0, col0, acc);

#pragma unroll
  for (int nt = 0; nt < 4; ++nt) {
    const int c  = col0 + nt * 16 + n;
    const int h  = c >> 6;
    const int d  = c & 63;
    const float bv_ = bias[c];
#pragma unroll
    for (int r = 0; r < 8; ++r) {
      const int grow = row0 + r + mb;        // = b*T + t
      const int b_   = grow >> 11;
      const int t    = grow & (TT - 1);
      const _Float16 val = (_Float16)(acc[nt][r] + bv_);
      if (blockIdx.z == 0)
        Qo[(((size_t)(b_ * HH + h)) * TT + t) * DD + d] = val;
      else if (blockIdx.z == 1)
        Ko[(((size_t)(b_ * HH + h)) * TT + t) * DD + d] = val;
      else
        Vto[(((size_t)(b_ * HH + h)) * DD + d) * TT + t] = val;
    }
  }
}

// ---------------------------------------------------------------------------
// Kernel: causal flash attention.  8 waves/block, one wave per 16-row Q tile.
// Q,K f16 [B,H,T,D]; V f16 [B,H,D,T]; Y f16 [B,T,C].
// All B-operand loads are contiguous 16-half runs.
// grid = (TT/128, BB*HH), block = 256
// ---------------------------------------------------------------------------
__global__ __launch_bounds__(256)
void attn_kernel(const _Float16* __restrict__ Q, const _Float16* __restrict__ K,
                 const _Float16* __restrict__ Vt, _Float16* __restrict__ Y) {
  __shared__ _Float16 pshared[8][16 * 32];

  const int wave = threadIdx.x >> 5;
  const int lane = threadIdx.x & 31;
  _Float16* pbuf = &pshared[wave][0];

  const int t0 = (blockIdx.x * 8 + wave) * 16;
  const int bh = blockIdx.y;
  const int b_ = bh >> 4;
  const int h  = bh & 15;
  const _Float16* q = Q  + (size_t)bh * TT * DD;
  const _Float16* k = K  + (size_t)bh * TT * DD;
  const _Float16* v = Vt + (size_t)bh * DD * TT;

  const int m  = lane & 15;
  const int kh = lane >> 4;
  const int n  = lane & 15;
  const int mb = 8 * kh;

  // Q tile in A-layout, two K-chunks (d 0..31, 32..63)
  v16h qa[2];
  {
    const _Float16* qp = q + (size_t)(t0 + m) * DD;
#pragma unroll
    for (int c = 0; c < 2; ++c) {
      v16h a;
#pragma unroll
      for (int i = 0; i < 8; ++i) a[i]     = qp[c * 32 + kh * 16 + i];
#pragma unroll
      for (int i = 0; i < 8; ++i) a[8 + i] = qp[c * 32 + kh * 16 + 8 + i];
      qa[c] = a;
    }
  }

  v8f O0 = {}, O1 = {}, O2 = {}, O3 = {};
  float mrow[8], lrow[8];
#pragma unroll
  for (int r = 0; r < 8; ++r) { mrow[r] = -1e30f; lrow[r] = 0.0f; }

  const int nchunk = (t0 + 15) / 32 + 1;  // 32-wide causal key chunks
  for (int ch = 0; ch < nchunk; ++ch) {
    const int s0 = ch * 32;

    // scores: two 16x16 tiles (cols s0..s0+15, s0+16..s0+31)
    v8f sc[2];
#pragma unroll
    for (int hf = 0; hf < 2; ++hf) {
      v8f acc = {};
#pragma unroll
      for (int kc = 0; kc < 2; ++kc) {
        v16h bm;  // B = K^T: n-lane = key index s; 16 contiguous d per lane
        const _Float16* kp =
            k + (size_t)(s0 + hf * 16 + n) * DD + kc * 32 + kh * 16;
#pragma unroll
        for (int i = 0; i < 16; ++i) bm[i] = kp[i];
        acc = wmma16(qa[kc], bm, acc);
      }
      sc[hf] = acc;
    }

    // scale 1/sqrt(64), causal mask, online softmax update
    float p0[8], p1[8], alpha[8];
#pragma unroll
    for (int r = 0; r < 8; ++r) {
      const int trow = t0 + r + mb;
      const int s    = s0 + n;
      float x0 = sc[0][r] * 0.125f;
      float x1 = sc[1][r] * 0.125f;
      if (s > trow)      x0 = -10000.0f;
      if (s + 16 > trow) x1 = -10000.0f;
      const float cm   = half16_max(fmaxf(x0, x1));
      const float newm = fmaxf(mrow[r], cm);
      alpha[r] = __expf(mrow[r] - newm);
      p0[r]    = __expf(x0 - newm);
      p1[r]    = __expf(x1 - newm);
      lrow[r]  = lrow[r] * alpha[r] + half16_sum(p0[r] + p1[r]);
      mrow[r]  = newm;
    }
#pragma unroll
    for (int r = 0; r < 8; ++r) {
      O0[r] *= alpha[r]; O1[r] *= alpha[r]; O2[r] *= alpha[r]; O3[r] *= alpha[r];
    }

    // P: C-layout -> LDS row-major [16][32] -> A-layout (same-wave DS in-order)
#pragma unroll
    for (int r = 0; r < 8; ++r) {
      pbuf[(r + mb) * 32 + n]      = (_Float16)p0[r];
      pbuf[(r + mb) * 32 + 16 + n] = (_Float16)p1[r];
    }
    v16h pa;
#pragma unroll
    for (int i = 0; i < 8; ++i) pa[i]     = pbuf[m * 32 + kh * 16 + i];
#pragma unroll
    for (int i = 0; i < 8; ++i) pa[8 + i] = pbuf[m * 32 + kh * 16 + 8 + i];

    // O += P (16x32) @ V (32x64); Vt gives contiguous s-runs per lane
#pragma unroll
    for (int dt = 0; dt < 4; ++dt) {
      v16h bm;
      const _Float16* vp =
          v + (size_t)(dt * 16 + n) * TT + s0 + kh * 16;
#pragma unroll
      for (int i = 0; i < 16; ++i) bm[i] = vp[i];
      if (dt == 0)      O0 = wmma16(pa, bm, O0);
      else if (dt == 1) O1 = wmma16(pa, bm, O1);
      else if (dt == 2) O2 = wmma16(pa, bm, O2);
      else              O3 = wmma16(pa, bm, O3);
    }
  }

  // normalize and store Y in [B, T, C] f16 layout
#pragma unroll
  for (int r = 0; r < 8; ++r) {
    const float rl = 1.0f / lrow[r];
    const size_t base = ((size_t)(b_ * TT + t0 + r + mb)) * CC + h * DD + n;
    Y[base + 0]  = (_Float16)(O0[r] * rl);
    Y[base + 16] = (_Float16)(O1[r] * rl);
    Y[base + 32] = (_Float16)(O2[r] * rl);
    Y[base + 48] = (_Float16)(O3[r] * rl);
  }
}

// ---------------------------------------------------------------------------
// Kernel: output projection  out = Y(f16) @ Wp + bp  (f32 out)
// grid = (CC/64, ROWS/128), block = 256
// ---------------------------------------------------------------------------
__global__ __launch_bounds__(256)
void out_proj_kernel(const _Float16* __restrict__ Y,
                     const _Float16* __restrict__ Wtp,
                     const float* __restrict__ bp, float* __restrict__ out) {
  __shared__ _Float16 btile[2 * PANEL];
  const int wave = threadIdx.x >> 5;
  const int lane = threadIdx.x & 31;
  const int row0 = blockIdx.y * 128 + wave * 16;
  const int col0 = blockIdx.x * 64;
  const int kh = lane >> 4;
  const int n  = lane & 15;
  const int mb = 8 * kh;

  v8f acc[4] = {{}, {}, {}, {}};
  gemm_core(Y, Wtp, btile, row0, col0, acc);

#pragma unroll
  for (int nt = 0; nt < 4; ++nt) {
    const int c = col0 + nt * 16 + n;
    const float bv_ = bp[c];
#pragma unroll
    for (int r = 0; r < 8; ++r)
      out[(size_t)(row0 + r + mb) * CC + c] = acc[nt][r] + bv_;
  }
}

// ---------------------------------------------------------------------------
extern "C" void kernel_launch(void* const* d_in, const int* in_sizes, int n_in,
                              void* d_out, int out_size, void* d_ws, size_t ws_size,
                              hipStream_t stream) {
  const float* x  = (const float*)d_in[0];
  const float* Wk = (const float*)d_in[1];
  const float* bk = (const float*)d_in[2];
  const float* Wq = (const float*)d_in[3];
  const float* bq = (const float*)d_in[4];
  const float* Wv = (const float*)d_in[5];
  const float* bv = (const float*)d_in[6];
  const float* Wp = (const float*)d_in[7];
  const float* bp = (const float*)d_in[8];
  float* out = (float*)d_out;

  // workspace (all f16):
  //   xh [ROWS,CC] 16MB | Wtq/Wtk/Wtv/Wtp [CC,CC] 2MB each
  //   Q,K [B,H,T,D] 16MB each | Vt [B,H,D,T] 16MB | Y [B,T,C] 16MB
  _Float16* xh  = (_Float16*)d_ws;
  _Float16* Wtq = xh  + (size_t)ROWS * CC;
  _Float16* Wtk = Wtq + (size_t)CC * CC;
  _Float16* Wtv = Wtk + (size_t)CC * CC;
  _Float16* Wtp = Wtv + (size_t)CC * CC;
  _Float16* Qw  = Wtp + (size_t)CC * CC;
  _Float16* Kw  = Qw  + (size_t)ROWS * CC;
  _Float16* Vtw = Kw  + (size_t)ROWS * CC;
  _Float16* Yw  = Vtw + (size_t)ROWS * CC;

  cvt_x_kernel<<<(ROWS * CC / 4) / 256, 256, 0, stream>>>(x, xh);
  cvt_wt_kernel<<<dim3(CC / 32, CC / 32, 4), 256, 0, stream>>>(
      Wq, Wk, Wv, Wp, Wtq, Wtk, Wtv, Wtp);
  qkv_proj_kernel<<<dim3(CC / 64, ROWS / 128, 3), 256, 0, stream>>>(
      xh, Wtq, bq, Wtk, bk, Wtv, bv, Qw, Kw, Vtw);
  attn_kernel<<<dim3(TT / 128, BB * HH), 256, 0, stream>>>(Qw, Kw, Vtw, Yw);
  out_proj_kernel<<<dim3(CC / 64, ROWS / 128), 256, 0, stream>>>(
      Yw, Wtp, bp, out);
}
